// MultiHeadAttention_80762565034040
// MI455X (gfx1250) — matrix operand order repaired and checked by reading, not compiled, not run on hardware
//
#include <hip/hip_runtime.h>
#include <hip/hip_bf16.h>

// ---------------------------------------------------------------------------
// Linear multi-head attention for MI455X (gfx1250), wave32 + WMMA bf16.
//   q/k/v proj: [32768,512]x[512,512] bf16 WMMA GEMM (double-buffered LDS)
//   k col-softmax folded into scores = diag(1/colsum) * expK^T V / 8
//   q row-softmax fused into Q-projection epilogue
//   out = softQ @ scores per head, final att @ Wo WMMA GEMM (f32 out)
// Raw global->LDS tile copies use GLOBAL_LOAD_ASYNC_TO_LDS_B128 (ASYNCcnt).
// ---------------------------------------------------------------------------

typedef __bf16 bf16;
typedef __attribute__((ext_vector_type(16))) __bf16 v16bf;
typedef __attribute__((ext_vector_type(8)))  __bf16 v8bf;
typedef __attribute__((ext_vector_type(4)))  __bf16 v4bf;
typedef __attribute__((ext_vector_type(8)))  float  v8f;

#define BM 128
#define BN 64
#define BK 32
#define LSTR 48   // LDS row stride (bf16 elems): 96B rows -> 16B/32B aligned frags

__device__ __forceinline__ v8f wmma_bf16(v16bf a, v16bf b, v8f c) {
  // D = A(16x32 bf16) * B(32x16 bf16) + C(16x16 f32)
  return __builtin_amdgcn_wmma_f32_16x16x32_bf16(false, a, false, b,
                                                 (short)0, c, false, false);
}

// Async contiguous 16B global->LDS copy; completion tracked by ASYNCcnt.
__device__ __forceinline__ void async_copy_b128(const void* gptr, void* lptr) {
  unsigned lds = (unsigned)(unsigned long long)lptr;   // LDS address: low 32 bits of the pointer
  asm volatile("global_load_async_to_lds_b128 %0, %1, off"
               :: "v"(lds), "v"(gptr) : "memory");
}
__device__ __forceinline__ void wait_async() {
  asm volatile("s_wait_asynccnt 0x0" ::: "memory");
}

// A fragment (16x32 bf16): lane l<16 -> row M=l, K 0..15; l>=16 -> K 16..31
__device__ __forceinline__ v16bf load_a_frag(const bf16* tile, int stride, int lane) {
  int lm = lane & 15, lh = lane >> 4;
  const bf16* p = tile + lm * stride + lh * 16;
  v8bf lo = *(const v8bf*)p;
  v8bf hi = *(const v8bf*)(p + 8);
  return __builtin_shufflevector(lo, hi, 0,1,2,3,4,5,6,7,8,9,10,11,12,13,14,15);
}

// B fragment (32x16 bf16) from [n][k]-major LDS tile: lane l holds col n=l&15,
// k = (l>>4)*16 .. +15 contiguous -> single 32B LDS read.
__device__ __forceinline__ v16bf load_b_frag(const bf16* tile, int stride, int lane) {
  int ln = lane & 15, lh = lane >> 4;
  return *(const v16bf*)(tile + ln * stride + lh * 16);
}

__device__ __forceinline__ v8f zero8() {
  v8f z;
#pragma unroll
  for (int i = 0; i < 8; ++i) z[i] = 0.0f;
  return z;
}

// ---------------------------------------------------------------------------
// Weight transpose + f32->bf16:  Wt[n][k] = (bf16)W[k][n]   (512x512)
// ---------------------------------------------------------------------------
__global__ void wt_kernel(const float* __restrict__ W, bf16* __restrict__ Wt) {
  int n = blockIdx.x;
  for (int k = threadIdx.x; k < 512; k += blockDim.x)
    Wt[(size_t)n * 512 + k] = (bf16)W[(size_t)k * 512 + n];
}

// ---------------------------------------------------------------------------
// Generic WMMA GEMM: C[M,N] = A[M,K] * Bt[N,K]^T
//   AMODE 0: A is f32 (convert to bf16 while staging); 1: A is bf16 (async copy)
//   EPI   0: store bf16   1: fused row-softmax over BN=64, store bf16   2: store f32
// Block 256 thr = 8 waves as 4(M) x 2(N); wave = 2x2 WMMA tiles (32x32).
// Double-buffered LDS (1 barrier / k-step) + global_prefetch 2 steps ahead.
// B tile (and bf16 A tile) staged with async global->LDS b128 copies.
// ---------------------------------------------------------------------------
template<int AMODE, int EPI>
__global__ void __launch_bounds__(256)
gemm_kernel(const void* __restrict__ Av, const bf16* __restrict__ Bt,
            void* __restrict__ Cv, int M, int N, int K) {
  // pool: As0|As1 (12288B each) + Bs0|Bs1 (6144B each) = 36864B.
  // EPI==1 overlays Ct (128*65*4 = 33280B) after the k-loop.
  __shared__ alignas(32) char smem[2 * BM * LSTR * 2 + 2 * BN * LSTR * 2];
  bf16* const As0 = (bf16*)smem;
  bf16* const As1 = As0 + BM * LSTR;
  bf16* const Bs0 = As1 + BM * LSTR;
  bf16* const Bs1 = Bs0 + BN * LSTR;
  float* const Ct = (float*)smem;

  const int tid  = threadIdx.x;
  const int lane = tid & 31;
  const int w    = tid >> 5;   // 0..7
  const int wm   = w >> 1;     // 0..3
  const int wn   = w & 1;      // 0..1
  const int n0   = blockIdx.x * BN;
  const int m0   = blockIdx.y * BM;

  v8f acc[2][2];
#pragma unroll
  for (int i = 0; i < 2; ++i)
#pragma unroll
    for (int j = 0; j < 2; ++j) acc[i][j] = zero8();

  const int arow = tid >> 1;            // 0..127
  const int acol = (tid & 1) * 16;      // 0|16
  const int brow = tid >> 2;            // 0..63
  const int bcol = (tid & 3) * 8;       // 0..24

  const float* Af = (const float*)Av;
  const bf16*  Ab = (const bf16*)Av;

  auto stage = [&](bf16* Asb, bf16* Bsb, int k0) {
    if (AMODE == 0) {
      const float* g = Af + (size_t)(m0 + arow) * K + k0 + acol;
#pragma unroll
      for (int i = 0; i < 4; ++i) {
        float4 v = *(const float4*)(g + 4 * i);
        v4bf b; b[0] = (bf16)v.x; b[1] = (bf16)v.y; b[2] = (bf16)v.z; b[3] = (bf16)v.w;
        *(v4bf*)&Asb[arow * LSTR + acol + 4 * i] = b;
      }
    } else {
      const bf16* g = Ab + (size_t)(m0 + arow) * K + k0 + acol;
#pragma unroll
      for (int i = 0; i < 2; ++i)
        async_copy_b128(g + 8 * i, &Asb[arow * LSTR + acol + 8 * i]);
    }
    const bf16* gb = Bt + (size_t)(n0 + brow) * K + k0 + bcol;
    async_copy_b128(gb, &Bsb[brow * LSTR + bcol]);
  };

  const int NIT = K / BK;
  stage(As0, Bs0, 0);
  wait_async();
  __syncthreads();

  for (int it = 0; it < NIT; ++it) {
    const int cur = it & 1;
    if (it + 2 < NIT) {   // pull tiles 2 steps ahead into cache (global_prefetch_b8)
      if (AMODE == 0)
        __builtin_prefetch(Af + (size_t)(m0 + arow) * K + (it + 2) * BK + acol, 0, 1);
      else
        __builtin_prefetch(Ab + (size_t)(m0 + arow) * K + (it + 2) * BK + acol, 0, 1);
      __builtin_prefetch(Bt + (size_t)(n0 + brow) * K + (it + 2) * BK + bcol, 0, 1);
    }
    if (it + 1 < NIT)     // fill the other buffer while this one is consumed
      stage(cur ? As0 : As1, cur ? Bs0 : Bs1, (it + 1) * BK);

    bf16* Ac = cur ? As1 : As0;
    bf16* Bc = cur ? Bs1 : Bs0;
    v16bf a0 = load_a_frag(&Ac[(wm * 32 +  0) * LSTR], LSTR, lane);
    v16bf a1 = load_a_frag(&Ac[(wm * 32 + 16) * LSTR], LSTR, lane);
    v16bf b0 = load_b_frag(&Bc[(wn * 32 +  0) * LSTR], LSTR, lane);
    v16bf b1 = load_b_frag(&Bc[(wn * 32 + 16) * LSTR], LSTR, lane);
    acc[0][0] = wmma_bf16(a0, b0, acc[0][0]);
    acc[0][1] = wmma_bf16(a0, b1, acc[0][1]);
    acc[1][0] = wmma_bf16(a1, b0, acc[1][0]);
    acc[1][1] = wmma_bf16(a1, b1, acc[1][1]);
    wait_async();          // next-buffer async copies landed in LDS
    __syncthreads();
  }

  // C/D layout: lane holds col n = lane&15; VGPR r -> row r + (lane>>4)*8
  const int lh8 = (lane >> 4) * 8;
  const int lnc = lane & 15;

  if (EPI == 0 || EPI == 2) {
#pragma unroll
    for (int ti = 0; ti < 2; ++ti)
#pragma unroll
      for (int tj = 0; tj < 2; ++tj) {
        int mr = m0 + wm * 32 + ti * 16 + lh8;
        int nc = n0 + wn * 32 + tj * 16 + lnc;
#pragma unroll
        for (int r = 0; r < 8; ++r) {
          if (EPI == 0) ((bf16*)Cv)[(size_t)(mr + r) * N + nc] = (bf16)acc[ti][tj][r];
          else          ((float*)Cv)[(size_t)(mr + r) * N + nc] = acc[ti][tj][r];
        }
      }
  } else {
    // fused q-softmax over the BN=64 head chunk (rowwise); Ct overlays tile LDS
#pragma unroll
    for (int ti = 0; ti < 2; ++ti)
#pragma unroll
      for (int tj = 0; tj < 2; ++tj) {
        int mr = wm * 32 + ti * 16 + lh8;
        int nc = wn * 32 + tj * 16 + lnc;
#pragma unroll
        for (int r = 0; r < 8; ++r) Ct[(mr + r) * (BN + 1) + nc] = acc[ti][tj][r];
      }
    __syncthreads();
    if (tid < BM) {
      float* row = &Ct[tid * (BN + 1)];
      float mx = -1e30f;
      for (int j = 0; j < BN; ++j) mx = fmaxf(mx, row[j]);
      float sum = 0.0f;
      for (int j = 0; j < BN; ++j) { float e = __expf(row[j] - mx); sum += e; row[j] = e; }
      float inv = 1.0f / sum;
      bf16* C = (bf16*)Cv;
      size_t base = (size_t)(m0 + tid) * N + n0;
#pragma unroll
      for (int j = 0; j < BN; j += 8) {
        v8bf o;
#pragma unroll
        for (int t = 0; t < 8; ++t) o[t] = (bf16)(row[j + t] * inv);
        *(v8bf*)&C[base + j] = o;
      }
    }
  }
}

// ---------------------------------------------------------------------------
// Per-(b,h,dk) online column max / sum(exp) over 8192 rows of the head view.
// Kp head view: contiguous [8192][64] at bh*8192*64.
// ---------------------------------------------------------------------------
__global__ void __launch_bounds__(256)
colstats_kernel(const bf16* __restrict__ Kp, float* __restrict__ colmax,
                float* __restrict__ colsum) {
  __shared__ float sm[4][64], ss[4][64];
  const int bh = blockIdx.x;
  const bf16* base = Kp + (size_t)bh * 8192 * 64;
  const int dk = threadIdx.x & 63;
  const int r0 = threadIdx.x >> 6;   // 0..3
  float m = -1e30f, s = 0.0f;
  for (int sr = r0; sr < 8192; sr += 4) {
    float x = (float)base[(size_t)sr * 64 + dk];
    float nm = fmaxf(m, x);
    s = s * __expf(m - nm) + __expf(x - nm);
    m = nm;
  }
  sm[r0][dk] = m; ss[r0][dk] = s;
  __syncthreads();
  if (threadIdx.x < 64) {
    float M = sm[0][dk], S = ss[0][dk];
#pragma unroll
    for (int i = 1; i < 4; ++i) {
      float nm = fmaxf(M, sm[i][dk]);
      S = S * __expf(M - nm) + ss[i][dk] * __expf(sm[i][dk] - nm);
      M = nm;
    }
    colmax[bh * 64 + dk] = M;
    colsum[bh * 64 + dk] = S;
  }
}

// ---------------------------------------------------------------------------
// scoresT[bh][dv][dk] = (sum_s exp(K[s,dk]-max) * V[s,dv]) / (colsum[dk]*8)
// One block per (b,h); 16 waves, one 16x16 WMMA tile each; K-dim = 8192.
// Double-buffered LDS staging (1 barrier / 32-row step) + prefetch.
// (exp + transpose staging must stay on the VALU/DS path.)
// ---------------------------------------------------------------------------
__global__ void __launch_bounds__(512)
scores_kernel(const bf16* __restrict__ Kp, const bf16* __restrict__ Vp,
              const float* __restrict__ colmax, const float* __restrict__ colsum,
              bf16* __restrict__ scoresT) {
  __shared__ alignas(32) bf16 Kt[2][64 * LSTR];   // [dk][s]
  __shared__ alignas(32) bf16 Vt[2][64 * LSTR];   // [dv][s]
  __shared__ float cmax_s[64], csum_s[64];

  const int bh = blockIdx.x;
  const bf16* kb = Kp + (size_t)bh * 8192 * 64;
  const bf16* vb = Vp + (size_t)bh * 8192 * 64;
  const int tid = threadIdx.x, lane = tid & 31, w = tid >> 5;  // 16 waves
  const int tm = w >> 2, tn = w & 3;
  if (tid < 64)  cmax_s[tid] = colmax[bh * 64 + tid];
  else if (tid < 128) csum_s[tid - 64] = colsum[bh * 64 + tid - 64];

  const int sl = tid >> 4;           // 0..31
  const int c4 = (tid & 15) * 4;     // 0..60

  auto stage = [&](int buf, int s0) {
    v4bf kv = *(const v4bf*)&kb[(size_t)(s0 + sl) * 64 + c4];
    v4bf vv = *(const v4bf*)&vb[(size_t)(s0 + sl) * 64 + c4];
#pragma unroll
    for (int i = 0; i < 4; ++i) {
      Kt[buf][(c4 + i) * LSTR + sl] = (bf16)__expf((float)kv[i] - cmax_s[c4 + i]);
      Vt[buf][(c4 + i) * LSTR + sl] = vv[i];
    }
  };

  __syncthreads();                 // cmax_s ready before first stage uses it
  v8f acc = zero8();
  stage(0, 0);
  __syncthreads();

  const int NIT = 8192 / 32;
  for (int it = 0; it < NIT; ++it) {
    const int cur = it & 1;
    if (it + 2 < NIT) {
      __builtin_prefetch(&kb[(size_t)((it + 2) * 32 + sl) * 64 + c4], 0, 1);
      __builtin_prefetch(&vb[(size_t)((it + 2) * 32 + sl) * 64 + c4], 0, 1);
    }
    if (it + 1 < NIT) stage(cur ^ 1, (it + 1) * 32);
    v16bf a = load_a_frag(&Kt[cur][(tm * 16) * LSTR], LSTR, lane);   // A = expK^T
    v16bf b = load_b_frag(&Vt[cur][(tn * 16) * LSTR], LSTR, lane);   // B = V
    acc = wmma_bf16(a, b, acc);
    __syncthreads();
  }

  const int lh = lane >> 4, ln = lane & 15;
  const int dv = tn * 16 + ln;
  const int dkb = tm * 16 + lh * 8;
  v8bf outv;
#pragma unroll
  for (int r = 0; r < 8; ++r)
    outv[r] = (bf16)(acc[r] / (csum_s[dkb + r] * 8.0f));    // TEMP = sqrt(64)
  *(v8bf*)&scoresT[(size_t)bh * 4096 + dv * 64 + dkb] = outv;
}

// ---------------------------------------------------------------------------
// att[b][s'][h*64+dv] = softQ_head[s',:] @ scores_head  (M=8192, N=64, K=64)
// Block: 128 rows; 8 waves x (1 Mtile x 4 Ntiles); K = 2 WMMA steps.
// Tile staging via async global->LDS b128 copies (ASYNCcnt).
// ---------------------------------------------------------------------------
__global__ void __launch_bounds__(256)
outhead_kernel(const bf16* __restrict__ softQ, const bf16* __restrict__ scoresT,
               bf16* __restrict__ att) {
  __shared__ alignas(32) bf16 Qs[128 * 64];   // [m][k]
  __shared__ alignas(32) bf16 Ss[64 * 64];    // [dv][dk]
  const int bh = blockIdx.y, b = bh >> 3, h = bh & 7;
  const int m0 = blockIdx.x * 128;
  const bf16* qb = softQ + (size_t)bh * 8192 * 64;
  const int tid = threadIdx.x, lane = tid & 31, w = tid >> 5;

  { // stage softQ tile 128x64 (async copies, contiguous layout match)
    int row = tid >> 1, ch = (tid & 1) * 32;
    const bf16* g = qb + (size_t)(m0 + row) * 64 + ch;
#pragma unroll
    for (int i = 0; i < 4; ++i)
      async_copy_b128(g + 8 * i, &Qs[row * 64 + ch + 8 * i]);
  }
  { // stage scoresT 64x64 (async)
    int row = tid >> 2, ch = (tid & 3) * 16;
    const bf16* g = scoresT + (size_t)bh * 4096 + row * 64 + ch;
    async_copy_b128(g,     &Ss[row * 64 + ch]);
    async_copy_b128(g + 8, &Ss[row * 64 + ch + 8]);
  }
  wait_async();
  __syncthreads();

  const int mt = w * 16;
  v16bf a0 = load_a_frag(&Qs[mt * 64 +  0], 64, lane);
  v16bf a1 = load_a_frag(&Qs[mt * 64 + 32], 64, lane);
  const int ln = lane & 15, lh = lane >> 4;

#pragma unroll
  for (int nt = 0; nt < 4; ++nt) {
    v16bf b0 = load_b_frag(&Ss[(nt * 16) * 64 +  0], 64, lane);
    v16bf b1 = load_b_frag(&Ss[(nt * 16) * 64 + 32], 64, lane);
    v8f acc = zero8();
    acc = wmma_bf16(a0, b0, acc);
    acc = wmma_bf16(a1, b1, acc);
    int dv = nt * 16 + ln;
    int mrow = m0 + mt + lh * 8;
    size_t base = ((size_t)b * 8192 + mrow) * 512 + h * 64 + dv;
#pragma unroll
    for (int r = 0; r < 8; ++r)
      att[base + (size_t)r * 512] = (bf16)acc[r];
  }
}

// ---------------------------------------------------------------------------
extern "C" void kernel_launch(void* const* d_in, const int* in_sizes, int n_in,
                              void* d_out, int out_size, void* d_ws, size_t ws_size,
                              hipStream_t stream) {
  const float* queries = (const float*)d_in[0];
  const float* keys    = (const float*)d_in[1];
  const float* values  = (const float*)d_in[2];
  const float* Wq = (const float*)d_in[3];
  const float* Wk = (const float*)d_in[4];
  const float* Wv = (const float*)d_in[5];
  const float* Wo = (const float*)d_in[6];
  float* out = (float*)d_out;

  const int M = 32768, D = 512;           // B*S rows, model dim
  const size_t WSZ = (size_t)512 * 512;   // one weight matrix
  const size_t PSZ = (size_t)M * 512;     // one projection buffer

  bf16* WqT = (bf16*)d_ws;
  bf16* WkT = WqT + WSZ;
  bf16* WvT = WkT + WSZ;
  bf16* WoT = WvT + WSZ;
  bf16* Kp  = WoT + WSZ;                  // bf16 [32768][512] == [B][8][8192][64]
  bf16* Vp  = Kp + PSZ;
  bf16* Qs  = Vp + PSZ;                   // softmaxed Q, bf16
  bf16* att = Qs + PSZ;                   // [B][8192][512] pre-Wo
  bf16* scT = att + PSZ;                  // scores^T per head: [32][64][64]
  float* colmax = (float*)(scT + (size_t)32 * 4096);
  float* colsum = colmax + 32 * 64;

  // 1) transpose + convert weights to bf16 [N][K]
  wt_kernel<<<dim3(512), dim3(256), 0, stream>>>(Wq, WqT);
  wt_kernel<<<dim3(512), dim3(256), 0, stream>>>(Wk, WkT);
  wt_kernel<<<dim3(512), dim3(256), 0, stream>>>(Wv, WvT);
  wt_kernel<<<dim3(512), dim3(256), 0, stream>>>(Wo, WoT);

  // 2) projections (WMMA bf16)
  dim3 gg(D / BN, M / BM);   // (8, 256)
  gemm_kernel<0, 0><<<gg, 256, 0, stream>>>(keys,    WkT, Kp, M, D, D);
  gemm_kernel<0, 0><<<gg, 256, 0, stream>>>(values,  WvT, Vp, M, D, D);
  gemm_kernel<0, 1><<<gg, 256, 0, stream>>>(queries, WqT, Qs, M, D, D);  // fused q-softmax

  // 3) column softmax stats for K (per b,h,dk over 8192)
  colstats_kernel<<<dim3(32), dim3(256), 0, stream>>>(Kp, colmax, colsum);

  // 4) scores = diag(1/colsum) * expK^T V / sqrt(64)   (WMMA, K-dim 8192)
  scores_kernel<<<dim3(32), dim3(512), 0, stream>>>(Kp, Vp, colmax, colsum, scT);

  // 5) out = softQ @ scores, written transposed into att (WMMA)
  outhead_kernel<<<dim3(64, 32), dim3(256), 0, stream>>>(Qs, scT, att);

  // 6) final: att @ Wo -> f32 output (WMMA)
  gemm_kernel<1, 2><<<gg, 256, 0, stream>>>(att, WoT, out, M, D, D);
}
